// GroupQueryAttention_25675314495666
// MI455X (gfx1250) — matrix operand order repaired and checked by reading, not compiled, not run on hardware
//
#include <hip/hip_runtime.h>

#define B_  2
#define T_  2048
#define H_  2048
#define NH_ 16
#define HD_ 128

typedef __attribute__((ext_vector_type(16))) _Float16 v16h;
typedef __attribute__((ext_vector_type(8)))  _Float16 v8h;
typedef __attribute__((ext_vector_type(8)))  float    v8f;

__device__ __forceinline__ v16h cat8(v8h a, v8h b) {
    return __builtin_shufflevector(a, b, 0,1,2,3,4,5,6,7,8,9,10,11,12,13,14,15);
}
__device__ __forceinline__ v8h cvt8(v8f x) {
    v8h r;
#pragma unroll
    for (int i = 0; i < 8; ++i) r[i] = (_Float16)x[i];
    return r;
}
// A fragment (16x32 f16, lane=row L%16): 16 contiguous halves at k+16*hf
__device__ __forceinline__ v16h loadA_f16(const _Float16* p, int hf) {
    return *(const v16h*)(p + 16 * hf);
}

// ---------------------------------------------------------------------------
// Bulk f32 -> f16 conversion (8 elements / thread, vectorized b128 traffic)
// ---------------------------------------------------------------------------
__global__ __launch_bounds__(256) void cvt_f16_kernel(
    const float* __restrict__ in, _Float16* __restrict__ out, int n8)
{
    int i = blockIdx.x * 256 + threadIdx.x;
    if (i < n8) {
        v8f x = *(const v8f*)(in + (size_t)i * 8);
        *(v8h*)(out + (size_t)i * 8) = cvt8(x);
    }
}

// ---------------------------------------------------------------------------
// out = (X @ W.T + bias) * outScale   with X, W already f16 (M = B_*T_, N = K = H_)
// One wave computes a 32(M) x 64(N) tile: 2 A-frags x 4 B-frags = 8 WMMA / k-chunk
// (~24 bytes of L2 traffic per WMMA).
// MODE 0: f16 out, scatter to [B][NH][T][HD]   (Q and K)
// MODE 1: f16 out, scatter to [B][NH][HD][T]   (V transposed)
// MODE 2: f32 out, row-major [M][H]            (final projection)
// ---------------------------------------------------------------------------
template <int MODE>
__global__ __launch_bounds__(32) void wmma_linear_kernel(
    const _Float16* __restrict__ X, const _Float16* __restrict__ W,
    const float* __restrict__ bias, void* __restrict__ outp, float outScale)
{
    const int lane = threadIdx.x & 31;
    const int col  = lane & 15;
    const int hf   = lane >> 4;
    const int m0   = blockIdx.x * 32;
    const int n0   = blockIdx.y * 64;

    v8f acc[2][4] = {};
    const _Float16* xrow0 = X + (size_t)(m0 + col) * H_;
    const _Float16* xrow1 = xrow0 + (size_t)16 * H_;

    for (int k0 = 0; k0 < H_; k0 += 32) {
        v16h a0 = loadA_f16(xrow0 + k0, hf);
        v16h a1 = loadA_f16(xrow1 + k0, hf);
#pragma unroll
        for (int nt = 0; nt < 4; ++nt) {
            // B fragment: lane = output col, 16 contiguous K values of row n of W
            v16h b = *(const v16h*)(W + (size_t)(n0 + nt * 16 + col) * H_ + k0 + 16 * hf);
            acc[0][nt] = __builtin_amdgcn_wmma_f32_16x16x32_f16(
                false, a0, false, b, (short)0, acc[0][nt], false, false);
            acc[1][nt] = __builtin_amdgcn_wmma_f32_16x16x32_f16(
                false, a1, false, b, (short)0, acc[1][nt], false, false);
        }
    }

#pragma unroll
    for (int mi = 0; mi < 2; ++mi) {
#pragma unroll
        for (int nt = 0; nt < 4; ++nt) {
            const int n = n0 + nt * 16 + col;
            const float bv = bias[n];
#pragma unroll
            for (int r = 0; r < 8; ++r) {
                float v = (acc[mi][nt][r] + bv) * outScale;
                const int mrow = m0 + mi * 16 + r + 8 * hf;  // D layout row
                if constexpr (MODE == 2) {
                    ((float*)outp)[(size_t)mrow * H_ + n] = v;
                } else {
                    const int bb = mrow >> 11, tt = mrow & (T_ - 1);
                    const int hh = n >> 7,    dd = n & (HD_ - 1);
                    _Float16* o = (_Float16*)outp;
                    if constexpr (MODE == 0)
                        o[(((size_t)bb * NH_ + hh) * T_ + tt) * HD_ + dd] = (_Float16)v;
                    else
                        o[(((size_t)bb * NH_ + hh) * HD_ + dd) * T_ + tt] = (_Float16)v;
                }
            }
        }
    }
}

// ---------------------------------------------------------------------------
// Flash attention: one wave handles 16 query rows of one (b,h); online softmax
// over 32-key chunks.  Q pre-scaled by HD^-0.5 in the projection kernel.
// ---------------------------------------------------------------------------
__global__ __launch_bounds__(128) void flash_attn_kernel(
    const _Float16* __restrict__ Qh, const _Float16* __restrict__ Kh,
    const _Float16* __restrict__ Vt, const float* __restrict__ mask,
    _Float16* __restrict__ attnOut)
{
    __shared__ _Float16 lds[4][16 * 32];   // per-wave private P tile (1 KB each)
    const int wave = threadIdx.x >> 5;
    const int lane = threadIdx.x & 31;
    const int col  = lane & 15;
    const int hf   = lane >> 4;
    const int bh   = blockIdx.y;           // b*NH + h
    const int b    = bh >> 4;
    const int h    = bh & (NH_ - 1);
    const int m0   = (blockIdx.x * 4 + wave) * 16;

    // Q fragments for the 16-row tile (HD=128 -> 4 K-chunks of 32)
    const _Float16* qrow = Qh + ((size_t)bh * T_ + m0 + col) * HD_;
    v16h aq[4];
#pragma unroll
    for (int kk = 0; kk < 4; ++kk) aq[kk] = loadA_f16(qrow + kk * 32, hf);

    v8f acc[8] = {};
    float mrun[8], lrun[8];
#pragma unroll
    for (int r = 0; r < 8; ++r) { mrun[r] = -1e30f; lrun[r] = 0.f; }

    const _Float16* Kbase = Kh + (size_t)bh * T_ * HD_;
    const _Float16* Vbase = Vt + (size_t)bh * HD_ * T_;
    const float*    mbase = mask + (size_t)b * T_ * T_;
    const float LOG2E = 1.4426950408889634f;

    for (int s0 = 0; s0 < T_; s0 += 32) {
        // ---- scores: two 16x16 tiles (keys s0..s0+15 and s0+16..s0+31) ----
        v8f sc0 = {}, sc1 = {};
#pragma unroll
        for (int kk = 0; kk < 4; ++kk) {
            v16h bk = *(const v16h*)(Kbase + (size_t)(s0 + col) * HD_ + kk * 32 + 16 * hf);
            sc0 = __builtin_amdgcn_wmma_f32_16x16x32_f16(false, aq[kk], false, bk,
                                                         (short)0, sc0, false, false);
        }
#pragma unroll
        for (int kk = 0; kk < 4; ++kk) {
            v16h bk = *(const v16h*)(Kbase + (size_t)(s0 + 16 + col) * HD_ + kk * 32 + 16 * hf);
            sc1 = __builtin_amdgcn_wmma_f32_16x16x32_f16(false, aq[kk], false, bk,
                                                         (short)0, sc1, false, false);
        }

        // ---- online softmax (row = r + 8*hf, spread over 16 lanes) ----
        float p0[8], p1[8], alpha[8];
#pragma unroll
        for (int r = 0; r < 8; ++r) {
            const float* mrow = mbase + (size_t)(m0 + r + 8 * hf) * T_ + s0;
            float s0v = (sc0[r] + mrow[col])      * LOG2E;
            float s1v = (sc1[r] + mrow[16 + col]) * LOG2E;
            float mx = fmaxf(s0v, s1v);
            mx = fmaxf(mx, __shfl_xor(mx, 1, 32));
            mx = fmaxf(mx, __shfl_xor(mx, 2, 32));
            mx = fmaxf(mx, __shfl_xor(mx, 4, 32));
            mx = fmaxf(mx, __shfl_xor(mx, 8, 32));
            float nm = fmaxf(mrun[r], mx);
            alpha[r] = exp2f(mrun[r] - nm);
            p0[r] = exp2f(s0v - nm);
            p1[r] = exp2f(s1v - nm);
            float ps = p0[r] + p1[r];
            ps += __shfl_xor(ps, 1, 32);
            ps += __shfl_xor(ps, 2, 32);
            ps += __shfl_xor(ps, 4, 32);
            ps += __shfl_xor(ps, 8, 32);
            lrun[r] = lrun[r] * alpha[r] + ps;
            mrun[r] = nm;
        }
#pragma unroll
        for (int n = 0; n < 8; ++n) {
#pragma unroll
            for (int r = 0; r < 8; ++r) acc[n][r] *= alpha[r];
        }

        // ---- redistribute P: D-layout -> A-layout via per-wave LDS tile ----
        _Float16* pl = lds[wave];
#pragma unroll
        for (int r = 0; r < 8; ++r) {
            pl[(r + 8 * hf) * 32 + col]      = (_Float16)p0[r];
            pl[(r + 8 * hf) * 32 + 16 + col] = (_Float16)p1[r];
        }
        __syncthreads();
        v16h ap = loadA_f16(pl + col * 32, hf);
        __syncthreads();

        // ---- P @ V  (Vt rows are contiguous B fragments) ----
#pragma unroll
        for (int n = 0; n < 8; ++n) {
            v16h bv = *(const v16h*)(Vbase + (size_t)(n * 16 + col) * T_ + s0 + 16 * hf);
            acc[n] = __builtin_amdgcn_wmma_f32_16x16x32_f16(false, ap, false, bv,
                                                            (short)0, acc[n], false, false);
        }
    }

    // ---- normalize and store attn as f16 [B][T][H] row-major ----
#pragma unroll
    for (int n = 0; n < 8; ++n) {
#pragma unroll
        for (int r = 0; r < 8; ++r) {
            float v = acc[n][r] / lrun[r];
            size_t row = (size_t)b * T_ + m0 + r + 8 * hf;
            attnOut[row * H_ + (size_t)h * HD_ + n * 16 + col] = (_Float16)v;
        }
    }
}

extern "C" void kernel_launch(void* const* d_in, const int* in_sizes, int n_in,
                              void* d_out, int out_size, void* d_ws, size_t ws_size,
                              hipStream_t stream)
{
    (void)in_sizes; (void)n_in; (void)out_size; (void)ws_size;
    const float* hs   = (const float*)d_in[0];
    const float* mask = (const float*)d_in[1];
    const float* Wq   = (const float*)d_in[2];
    const float* bq   = (const float*)d_in[3];
    const float* Wk   = (const float*)d_in[4];
    const float* bk   = (const float*)d_in[5];
    const float* Wv   = (const float*)d_in[6];
    const float* bv   = (const float*)d_in[7];
    const float* Wo   = (const float*)d_in[8];
    const float* bo   = (const float*)d_in[9];

    // workspace layout (f16 elements)
    const size_t szA = (size_t)B_ * NH_ * T_ * HD_;   // 8.39M  (Q/K/Vt/attn)
    const size_t szX = (size_t)B_ * T_ * H_;          // 8.39M  (hidden f16)
    const size_t szW = (size_t)H_ * H_;               // 4.19M  (each weight f16)
    _Float16* Qw  = (_Float16*)d_ws;
    _Float16* Kw  = Qw  + szA;
    _Float16* Vw  = Kw  + szA;
    _Float16* Aw  = Vw  + szA;
    _Float16* Xh  = Aw  + szA;
    _Float16* Wqh = Xh  + szX;
    _Float16* Wkh = Wqh + szW;
    _Float16* Wvh = Wkh + szW;
    _Float16* Woh = Wvh + szW;

    // one-shot f32 -> f16 packs (memory-bound, ~4 us total at 23.3 TB/s)
    const int nX8 = (int)(szX / 8), nW8 = (int)(szW / 8);
    cvt_f16_kernel<<<(nX8 + 255) / 256, 256, 0, stream>>>(hs, Xh,  nX8);
    cvt_f16_kernel<<<(nW8 + 255) / 256, 256, 0, stream>>>(Wq, Wqh, nW8);
    cvt_f16_kernel<<<(nW8 + 255) / 256, 256, 0, stream>>>(Wk, Wkh, nW8);
    cvt_f16_kernel<<<(nW8 + 255) / 256, 256, 0, stream>>>(Wv, Wvh, nW8);
    cvt_f16_kernel<<<(nW8 + 255) / 256, 256, 0, stream>>>(Wo, Woh, nW8);

    const dim3 ggrid(B_ * T_ / 32, H_ / 64);   // (128, 32)
    const dim3 gblk(32);
    const float scale = 0.08838834764831845f;  // HD^-0.5, folded into Q

    wmma_linear_kernel<0><<<ggrid, gblk, 0, stream>>>(Xh, Wqh, bq, Qw, scale);
    wmma_linear_kernel<0><<<ggrid, gblk, 0, stream>>>(Xh, Wkh, bk, Kw, 1.f);
    wmma_linear_kernel<1><<<ggrid, gblk, 0, stream>>>(Xh, Wvh, bv, Vw, 1.f);

    flash_attn_kernel<<<dim3(T_ / 64, B_ * NH_), dim3(128), 0, stream>>>(
        Qw, Kw, Vw, mask, Aw);

    wmma_linear_kernel<2><<<ggrid, gblk, 0, stream>>>(Aw, Woh, bo, d_out, 1.f);
}
